// HGRNCell_644245094456
// MI455X (gfx1250) — matrix operand order repaired and checked by reading, not compiled, not run on hardware
//
#include <hip/hip_runtime.h>
#include <hip/hip_bf16.h>

typedef unsigned short u16;
typedef __attribute__((ext_vector_type(16))) __bf16 v16bf;
typedef __attribute__((ext_vector_type(8)))  float  v8f;
typedef int v4i_vs __attribute__((vector_size(16)));   // matches builtin param type

union Frag { v16bf v; uint4 u[2]; };

#define D_DIM 1024
#define T_DIM 2048
#define B_DIM 8
#define M_DIM (B_DIM * T_DIM)   // 16384
#define N_DIM (3 * D_DIM)       // 3072
#define K_DIM D_DIM             // 1024

#define TILE_M 128
#define TILE_N 64
#define TILE_K 32
#define LDSS   40               // padded LDS row stride (ushorts) to spread banks

#define AS1 __attribute__((address_space(1)))
#define AS3 __attribute__((address_space(3)))

#if defined(__AMDGCN__) && __has_builtin(__builtin_amdgcn_global_load_async_to_lds_b128)
#define HAVE_ASYNC_LDS 1
#endif

__device__ __forceinline__ u16 f2bf(float f) {
    unsigned u = __float_as_uint(f);
    unsigned r = u + 0x7FFFu + ((u >> 16) & 1u);   // round-to-nearest-even
    return (u16)(r >> 16);
}
__device__ __forceinline__ float bf2f(u16 h) {
    return __uint_as_float(((unsigned)h) << 16);
}

// 16B global -> LDS copy; async (ASYNCcnt-tracked, no VGPR staging) if available
__device__ __forceinline__ void cp16(const u16* g, u16* l) {
#ifdef HAVE_ASYNC_LDS
    // generic->AS3 is "low 32 bits" on amdgcn (flat LDS aperture rule)
    __builtin_amdgcn_global_load_async_to_lds_b128(
        (AS1 v4i_vs*)(uintptr_t)g,
        (AS3 v4i_vs*)(unsigned)(uintptr_t)l,
        0, 0);
#else
    *(uint4*)l = *(const uint4*)g;
#endif
}
__device__ __forceinline__ void wait_async_lds() {
#ifdef HAVE_ASYNC_LDS
#if __has_builtin(__builtin_amdgcn_s_wait_asynccnt)
    __builtin_amdgcn_s_wait_asynccnt(0);
#else
    asm volatile("s_wait_asynccnt 0x0" ::: "memory");
#endif
#endif
}

// ---------------- Kernel 1: LayerNorm + bf16 hi/lo split --------------------
__global__ __launch_bounds__(256)
void ln_split_kernel(const float* __restrict__ x,
                     const float* __restrict__ gamma,
                     const float* __restrict__ beta,
                     u16* __restrict__ xh, u16* __restrict__ xl) {
    const int row  = blockIdx.x;                 // 0..16383
    const int tid  = threadIdx.x;
    const int lane = tid & 31;
    const int wave = tid >> 5;
    const float* xr = x + (size_t)row * D_DIM;

    float v[4], sum = 0.f, sq = 0.f;
#pragma unroll
    for (int i = 0; i < 4; ++i) {
        v[i] = xr[tid + i * 256];
        sum += v[i];
        sq  += v[i] * v[i];
    }
#pragma unroll
    for (int off = 16; off > 0; off >>= 1) {
        sum += __shfl_down(sum, off, 32);
        sq  += __shfl_down(sq,  off, 32);
    }
    __shared__ float red[16];
    if (lane == 0) { red[wave] = sum; red[8 + wave] = sq; }
    __syncthreads();
    float tot = 0.f, tsq = 0.f;
#pragma unroll
    for (int w = 0; w < 8; ++w) { tot += red[w]; tsq += red[8 + w]; }
    const float mean = tot * (1.f / D_DIM);
    const float var  = tsq * (1.f / D_DIM) - mean * mean;
    const float inv  = rsqrtf(var + 1e-5f);

#pragma unroll
    for (int i = 0; i < 4; ++i) {
        const int idx = tid + i * 256;
        float xn = (v[i] - mean) * inv * gamma[idx] + beta[idx];
        u16 hi = f2bf(xn);
        u16 lo = f2bf(xn - bf2f(hi));
        xh[(size_t)row * D_DIM + idx] = hi;
        xl[(size_t)row * D_DIM + idx] = lo;
    }
}

// ---------------- Kernel 2: split W into bf16 hi/lo -------------------------
__global__ __launch_bounds__(256)
void wsplit_kernel(const float* __restrict__ W,
                   u16* __restrict__ Wh, u16* __restrict__ Wl, int n) {
    int i = blockIdx.x * blockDim.x + threadIdx.x;
    if (i < n) {
        float w = W[i];
        u16 hi = f2bf(w);
        Wh[i] = hi;
        Wl[i] = f2bf(w - bf2f(hi));
    }
}

// ---------------- Kernel 3: bf16x3 WMMA GEMM, async double-buffered ---------
// A = xn [M,K] row-major (hi/lo), B = W [N,K] row-major (hi/lo) == native
// B-fragment layout (N-major, K-contiguous). Block tile: 128(M) x 64(N).
// Steady-state loop stages unconditionally; last K-step is peeled so the hot
// loop is straight-line: async-stage -> ds frag loads -> 12 WMMA -> wait+bar.
__global__ __launch_bounds__(256)
void gemm_bf16x3_kernel(const u16* __restrict__ Ah_g, const u16* __restrict__ Al_g,
                        const u16* __restrict__ Bh_g, const u16* __restrict__ Bl_g,
                        const float* __restrict__ bias, float* __restrict__ Cout) {
    __shared__ u16 sA[2][2][TILE_M * LDSS];   // [buf][hi/lo][m][k]  2x20KB
    __shared__ u16 sB[2][2][TILE_N * LDSS];   // [buf][hi/lo][n][k]  2x10KB

    const int tid  = threadIdx.x;
    const int lane = tid & 31;
    const int wave = tid >> 5;
    const int wm   = wave & 3;             // 4 M-subtiles of 32
    const int wn   = wave >> 2;            // 2 N-subtiles of 32
    const int m0   = blockIdx.y * TILE_M;
    const int n0   = blockIdx.x * TILE_N;

    const int lm  = lane & 15;
    const int agk = (lane >> 4) * 8;       // A-frag K base per lane half
    const int bgk = (lane >> 4) * 8;       // B-frag K base per lane half

    // per-thread staging geometry (fixed): row = tid/4 (+64 for 2nd A chunk),
    // 16B column chunk q = tid%4
    const int srow = tid >> 2;
    const int sq   = (tid & 3) * 8;
    const int ldsA = srow * LDSS + sq;
    const int ldsB = srow * LDSS + sq;     // valid when srow < 64

    v8f acc[2][2];
#pragma unroll
    for (int i = 0; i < 2; ++i)
#pragma unroll
        for (int j = 0; j < 2; ++j)
#pragma unroll
            for (int r = 0; r < 8; ++r) acc[i][j][r] = 0.f;

    // stage tile k0 into buffer `buf` (async if available)
    auto stage = [&](int buf, int k0) {
        size_t ga = (size_t)(m0 + srow) * K_DIM + k0 + sq;
        cp16(&Ah_g[ga], &sA[buf][0][ldsA]);
        cp16(&Al_g[ga], &sA[buf][1][ldsA]);
        size_t ga2 = ga + (size_t)64 * K_DIM;
        cp16(&Ah_g[ga2], &sA[buf][0][ldsA + 64 * LDSS]);
        cp16(&Al_g[ga2], &sA[buf][1][ldsA + 64 * LDSS]);
        size_t gb = (size_t)(n0 + srow) * K_DIM + k0 + sq;
        cp16(&Bh_g[gb], &sB[buf][0][ldsB]);
        cp16(&Bl_g[gb], &sB[buf][1][ldsB]);
    };

    // one K-step of 12 WMMAs out of buffer `buf`
    auto compute = [&](int buf) {
        Frag fah[2], fal[2], fbh[2], fbl[2];
#pragma unroll
        for (int i = 0; i < 2; ++i) {
            // A 16x32 bf16 layout: lanes 0-15 row M=lane, K {0-7,16-23};
            //                      lanes 16-31 row M=lane-16, K {8-15,24-31}
            int mrow = (wm * 32 + i * 16 + lm) * LDSS;
            fah[i].u[0] = *(const uint4*)&sA[buf][0][mrow + agk];
            fah[i].u[1] = *(const uint4*)&sA[buf][0][mrow + agk + 16];
            fal[i].u[0] = *(const uint4*)&sA[buf][1][mrow + agk];
            fal[i].u[1] = *(const uint4*)&sA[buf][1][mrow + agk + 16];
            // B 32x16 bf16 layout: lanes 0-15 col N=lane, K {0-7,16-23};
            //                      lanes 16-31 col N=lane-16, K {8-15,24-31}
            int nrow = (wn * 32 + i * 16 + lm) * LDSS;
            fbh[i].u[0] = *(const uint4*)&sB[buf][0][nrow + bgk];
            fbh[i].u[1] = *(const uint4*)&sB[buf][0][nrow + bgk + 16];
            fbl[i].u[0] = *(const uint4*)&sB[buf][1][nrow + bgk];
            fbl[i].u[1] = *(const uint4*)&sB[buf][1][nrow + bgk + 16];
        }
#pragma unroll
        for (int i = 0; i < 2; ++i)
#pragma unroll
            for (int j = 0; j < 2; ++j) {
                acc[i][j] = __builtin_amdgcn_wmma_f32_16x16x32_bf16(
                    false, fah[i].v, false, fbh[j].v, (short)0, acc[i][j], false, false);
                acc[i][j] = __builtin_amdgcn_wmma_f32_16x16x32_bf16(
                    false, fah[i].v, false, fbl[j].v, (short)0, acc[i][j], false, false);
                acc[i][j] = __builtin_amdgcn_wmma_f32_16x16x32_bf16(
                    false, fal[i].v, false, fbh[j].v, (short)0, acc[i][j], false, false);
            }
    };

    stage(0, 0);
    wait_async_lds();
    __syncthreads();

    int cur = 0;
    for (int k0 = 0; k0 < K_DIM - TILE_K; k0 += TILE_K) {   // steady state
        stage(cur ^ 1, k0 + TILE_K);                        // overlap with WMMA
        compute(cur);
        wait_async_lds();     // next tile's async copies landed (overlapped)
        __syncthreads();      // everyone done reading cur / writing nxt
        cur ^= 1;
    }
    compute(cur);             // peeled last K-step (no staging)

    // Epilogue: C/D f32 layout: VGPR r -> lanes 0-15 M=r, lanes 16-31 M=r+8
    const bool gate = (n0 >= D_DIM);   // block-uniform: gate thirds get sigmoid
#pragma unroll
    for (int i = 0; i < 2; ++i) {
#pragma unroll
        for (int j = 0; j < 2; ++j) {
            int n = n0 + wn * 32 + j * 16 + lm;
            float bn = bias[n];
            int mbase = m0 + wm * 32 + i * 16 + ((lane >> 4) << 3);
#pragma unroll
            for (int r = 0; r < 8; ++r) {
                float v = acc[i][j][r] + bn;
                if (gate) v = 1.f / (1.f + __expf(-v));
                Cout[(size_t)(mbase + r) * N_DIM + n] = v;
            }
        }
    }
}

// ---------------- Kernel 4: gated recurrent scan over T ---------------------
// Gates are pre-sigmoided by the GEMM epilogue. Each thread owns 4 adjacent
// channels (float4) and keeps an 8-step register ring of prefetched inputs,
// so ~24 loads are in flight while the serial tanh chain advances.
#define SDEPTH 8
__global__ __launch_bounds__(128)
void scan_kernel(const float* __restrict__ proj,
                 const float* __restrict__ u,
                 float* __restrict__ out) {
    const int gid = blockIdx.x * 128 + threadIdx.x;   // 0..2047
    const int b  = gid >> 8;
    const int dq = gid & 255;                         // float4 channel group
    const float4 uv = ((const float4*)u)[dq];

    const size_t pstride = N_DIM / 4;                 // 768 float4 per t
    const size_t ostride = D_DIM / 4;                 // 256 float4 per t
    const float4* pc = (const float4*)(proj + (size_t)b * T_DIM * N_DIM) + dq;
    const float4* pz = pc + D_DIM / 4;
    const float4* pr = pc + 2 * (D_DIM / 4);
    float4* po = (float4*)(out + (size_t)b * T_DIM * D_DIM) + dq;

    float4 rc[SDEPTH], rz[SDEPTH], rr[SDEPTH];
#pragma unroll
    for (int j = 0; j < SDEPTH; ++j) {
        size_t off = (size_t)j * pstride;
        rc[j] = pc[off]; rz[j] = pz[off]; rr[j] = pr[off];
    }

    float s0 = 0.f, s1 = 0.f, s2 = 0.f, s3 = 0.f;
    for (int t0 = 0; t0 < T_DIM; t0 += SDEPTH) {
#pragma unroll
        for (int j = 0; j < SDEPTH; ++j) {
            float4 c = rc[j], z = rz[j], r = rr[j];
            const int tn = t0 + j + SDEPTH;
            if (tn < T_DIM) {                  // refill ring slot (deep prefetch)
                size_t off = (size_t)tn * pstride;
                rc[j] = pc[off]; rz[j] = pz[off]; rr[j] = pr[off];
            }
            float h0 = tanhf(c.x + r.x * (uv.x * s0));
            float h1 = tanhf(c.y + r.y * (uv.y * s1));
            float h2 = tanhf(c.z + r.z * (uv.z * s2));
            float h3 = tanhf(c.w + r.w * (uv.w * s3));
            s0 = z.x * s0 + (1.f - z.x) * h0;
            s1 = z.y * s1 + (1.f - z.y) * h1;
            s2 = z.z * s2 + (1.f - z.z) * h2;
            s3 = z.w * s3 + (1.f - z.w) * h3;
            po[(size_t)(t0 + j) * ostride] = make_float4(s0, s1, s2, s3);
        }
    }
}

// ---------------------------------------------------------------------------
extern "C" void kernel_launch(void* const* d_in, const int* in_sizes, int n_in,
                              void* d_out, int out_size, void* d_ws, size_t ws_size,
                              hipStream_t stream) {
    const float* x     = (const float*)d_in[0];   // [8,2048,1024]
    const float* u     = (const float*)d_in[1];   // [1024]
    const float* W     = (const float*)d_in[2];   // [3072,1024]
    const float* bvec  = (const float*)d_in[3];   // [3072]
    const float* gamma = (const float*)d_in[4];   // [1024]
    const float* beta  = (const float*)d_in[5];   // [1024]
    float* out = (float*)d_out;                   // [8,2048,1024]

    const size_t MK = (size_t)M_DIM * K_DIM;      // 16M elems
    const size_t NK = (size_t)N_DIM * K_DIM;      // 3M elems

    u16* xh = (u16*)d_ws;                         // 32 MB
    u16* xl = xh + MK;                            // 32 MB
    u16* Wh = xl + MK;                            //  6 MB
    u16* Wl = Wh + NK;                            //  6 MB
    float* proj = (float*)(Wl + NK);              // 192 MB

    ln_split_kernel<<<M_DIM, 256, 0, stream>>>(x, gamma, beta, xh, xl);

    const int wn = (int)NK;
    wsplit_kernel<<<(wn + 255) / 256, 256, 0, stream>>>(W, Wh, Wl, wn);

    dim3 ggrid(N_DIM / TILE_N, M_DIM / TILE_M);   // (48, 128)
    gemm_bf16x3_kernel<<<ggrid, 256, 0, stream>>>(xh, xl, Wh, Wl, bvec, proj);

    scan_kernel<<<(2048 + 127) / 128, 128, 0, stream>>>(proj, u, out);
}
